// StiffNeuralODE_15642270892204
// MI455X (gfx1250) — matrix operand order repaired and checked by reading, not compiled, not run on hardware
//
#include <hip/hip_runtime.h>

#define OBS 20
#define WIDTH 64
#define BATCHN 4096
#define MAX_STEPS 256
#define DTC 0.01f
#define NEWTONI 3
#define WAVES_PER_WG 8
#define BLOCKSZ 256

typedef _Float16 v16h __attribute__((ext_vector_type(16)));
typedef float    v8f  __attribute__((ext_vector_type(8)));

// ---------------- LDS layout (floats) ----------------
#define S_W0S 0                   // 64 x 36 (W0 * diag(1/scale), padded)
#define S_B0 (S_W0S + 64*36)
#define S_W1 (S_B0 + 64)          // 64 x 68
#define S_B1 (S_W1 + 64*68)
#define S_W2 (S_B1 + 64)
#define S_B2 (S_W2 + 64*68)
#define S_W3 (S_B2 + 64)
#define S_B3 (S_W3 + 64*68)
#define S_W4 (S_B3 + 64)          // 20 x 68
#define S_B4 (S_W4 + 20*68)
#define S_SC (S_B4 + 20)
#define S_INV (S_SC + 20)
#define S_F32_END (S_INV + 20)
#define FRAG_OFF_BYTES (((S_F32_END*4)+31)&~31)
// f16 WMMA A-fragments (contiguous-k 16-bit A order), sizes in halves
#define FR_W1 0                   // 4 mt x 2 kc
#define FR_W2 (FR_W1 + 8*512)
#define FR_W3 (FR_W2 + 8*512)
#define FR_W4 (FR_W3 + 8*512)     // 2 mt x 2 kc
#define FR_END (FR_W4 + 4*512)
#define WS_OFF_BYTES (FRAG_OFF_BYTES + FR_END*2)
// per-wave scratch (floats)
#define WS_PER_WAVE 848
#define WV_ACTA 0
#define WV_ACTB 64
#define WV_EP   128               // elu' factors, 4 layers x 64
#define WV_Y    384
#define WV_YP   404
#define WV_AUG  424               // augmented system, column-major [21][20]
#define LDS_BYTES (WS_OFF_BYTES + WAVES_PER_WG*WS_PER_WAVE*4)

__device__ __forceinline__ v8f zero8() {
  v8f z;
#pragma unroll
  for (int i = 0; i < 8; ++i) z[i] = 0.f;
  return z;
}

__device__ __forceinline__ v8f wmma16(v16h a, v16h b, v8f c) {
  return __builtin_amdgcn_wmma_f32_16x16x32_f16(false, a, false, b, (short)0, c,
                                                false, false);
}

// lane^16 half-wave swap via ds_swizzle SWAPX16 (and=0x1f, or=0, xor=0x10)
__device__ __forceinline__ float swap16(float x) {
  return __uint_as_float(
      (unsigned)__builtin_amdgcn_ds_swizzle((int)__float_as_uint(x), 0x401f));
}

// constant-lane broadcast via v_readlane_b32 (no LDS traffic)
__device__ __forceinline__ float rdlane(float v, int l) {
  return __uint_as_float(
      (unsigned)__builtin_amdgcn_readlane((int)__float_as_uint(v), l));
}

// Pack a row-major f32 weight matrix into f16 A-fragments in LDS.
// 16-bit A 16x32 layout: lane l -> row m = 16*mt + (l&15);
// low lanes: K = 32*kc + {0..15}; high lanes: K = 32*kc + {16..31}
__device__ void packA(_Float16* dst, const float* __restrict__ W, int outd, int ind,
                      int mtiles, int ktiles, int tid) {
  const int total = mtiles * ktiles * 512;
  for (int idx = tid; idx < total; idx += BLOCKSZ) {
    int e = idx & 15;
    int l = (idx >> 4) & 31;
    int tile = idx >> 9;
    int kc = tile % ktiles;
    int mt = tile / ktiles;
    int m = 16 * mt + (l & 15);
    int kb = 32 * kc + ((l < 16) ? 0 : 16);
    int k = kb + e;
    float v = (m < outd && k < ind) ? W[m * ind + k] : 0.f;
    dst[idx] = (_Float16)v;
  }
}

__global__ void __launch_bounds__(BLOCKSZ)
stiff_node_kernel(const float* __restrict__ y0,
                  const float* __restrict__ W0, const float* __restrict__ b0,
                  const float* __restrict__ W1, const float* __restrict__ b1,
                  const float* __restrict__ W2, const float* __restrict__ b2,
                  const float* __restrict__ W3, const float* __restrict__ b3,
                  const float* __restrict__ W4, const float* __restrict__ b4,
                  const float* __restrict__ scale, const int* __restrict__ nsp,
                  float* __restrict__ out) {
  extern __shared__ char smem[];
  float* sf = (float*)smem;
  _Float16* fr = (_Float16*)(smem + FRAG_OFF_BYTES);

  const int tid = threadIdx.x;
  const int lane = tid & 31;
  const int wv = tid >> 5;

  // ---- stage weights ----
  for (int i = tid; i < 64 * 36; i += BLOCKSZ) {
    int o = i / 36, k = i % 36;
    sf[S_W0S + i] = (k < OBS) ? (W0[o * OBS + k] / scale[k]) : 0.f;
  }
  for (int i = tid; i < 64 * 68; i += BLOCKSZ) { int o = i / 68, k = i % 68; sf[S_W1 + i] = (k < WIDTH) ? W1[o * WIDTH + k] : 0.f; }
  for (int i = tid; i < 64 * 68; i += BLOCKSZ) { int o = i / 68, k = i % 68; sf[S_W2 + i] = (k < WIDTH) ? W2[o * WIDTH + k] : 0.f; }
  for (int i = tid; i < 64 * 68; i += BLOCKSZ) { int o = i / 68, k = i % 68; sf[S_W3 + i] = (k < WIDTH) ? W3[o * WIDTH + k] : 0.f; }
  for (int i = tid; i < 20 * 68; i += BLOCKSZ) { int o = i / 68, k = i % 68; sf[S_W4 + i] = (k < WIDTH) ? W4[o * WIDTH + k] : 0.f; }
  if (tid < 64) { sf[S_B0 + tid] = b0[tid]; sf[S_B1 + tid] = b1[tid]; sf[S_B2 + tid] = b2[tid]; sf[S_B3 + tid] = b3[tid]; }
  if (tid < OBS) { sf[S_B4 + tid] = b4[tid]; float sc = scale[tid]; sf[S_SC + tid] = sc; sf[S_INV + tid] = 1.f / sc; }
  packA(fr + FR_W1, W1, 64, 64, 4, 2, tid);
  packA(fr + FR_W2, W2, 64, 64, 4, 2, tid);
  packA(fr + FR_W3, W3, 64, 64, 4, 2, tid);
  packA(fr + FR_W4, W4, OBS, 64, 2, 2, tid);
  __syncthreads();

  float* ws   = sf + WS_OFF_BYTES / 4 + wv * WS_PER_WAVE;
  float* actA = ws + WV_ACTA;
  float* actB = ws + WV_ACTB;
  float* ep   = ws + WV_EP;
  float* yv   = ws + WV_Y;
  float* yp   = ws + WV_YP;
  float* aug  = ws + WV_AUG;
  const float* sSc = sf + S_SC;

  const int traj = blockIdx.x * WAVES_PER_WG + wv;
  if (lane < OBS) {
    float v = y0[traj * OBS + lane];
    yv[lane] = v; yp[lane] = v;
    out[(size_t)traj * OBS + lane] = v;  // row 0 = y0
  }
  int nsteps = *nsp;
  if (nsteps > MAX_STEPS) nsteps = MAX_STEPS;

  const bool lo = (lane < 16);
  const int  nl = lane & 15;
  const int  roff = lo ? 0 : 8;

  // f32 dense 64->64 with ELU; writes activations + elu' factors
  auto dense64 = [&](int wOff, int bOff, const float* in, float* outp, float* epO) {
#pragma unroll
    for (int h = 0; h < 2; ++h) {
      int o = lane + 32 * h;
      float acc = sf[bOff + o];
      const float4* w4p = (const float4*)&sf[wOff + o * 68];
      const float4* in4 = (const float4*)in;
#pragma unroll
      for (int k = 0; k < 16; ++k) {
        float4 w = w4p[k], x = in4[k];
        acc = fmaf(w.x, x.x, acc); acc = fmaf(w.y, x.y, acc);
        acc = fmaf(w.z, x.z, acc); acc = fmaf(w.w, x.w, acc);
      }
      float e = __expf(acc);
      epO[o]  = acc > 0.f ? 1.f : e;
      outp[o] = acc > 0.f ? acc : (e - 1.f);
    }
  };

  // Build B fragments (f16 32x16, contiguous-k order) from D tiles with elu'-row scaling.
  auto buildB = [&](const float* epL, const v8f (&Dt)[4][2], v16h (&Bf)[2][2]) {
    float epv[4][8];
#pragma unroll
    for (int mt = 0; mt < 4; ++mt) {
#pragma unroll
      for (int r = 0; r < 8; ++r) epv[mt][r] = epL[16 * mt + roff + r];
    }
#pragma unroll
    for (int kc = 0; kc < 2; ++kc) {
#pragma unroll
      for (int nt = 0; nt < 2; ++nt) {
        v16h b;
#pragma unroll
        for (int e = 0; e < 8; ++e) {
          float s0 = epv[2 * kc][e]     * Dt[2 * kc][nt][e];
          float s1 = epv[2 * kc + 1][e] * Dt[2 * kc + 1][nt][e];
          float w0 = swap16(s0);
          float w1 = swap16(s1);
          b[e]     = (_Float16)(lo ? s0 : w1);
          b[e + 8] = (_Float16)(lo ? w0 : s1);
        }
        Bf[kc][nt] = b;
      }
    }
  };

#pragma unroll 1
  for (int t = 0; t < nsteps; ++t) {
#pragma unroll 1
    for (int itn = 0; itn < NEWTONI; ++itn) {
      // ================= forward pass (f32, exact residual) =================
#pragma unroll
      for (int h = 0; h < 2; ++h) {
        int o = lane + 32 * h;
        float acc = sf[S_B0 + o];
        const float4* w4p = (const float4*)&sf[S_W0S + o * 36];
        const float4* in4 = (const float4*)yv;
#pragma unroll
        for (int k = 0; k < 5; ++k) {
          float4 w = w4p[k], x = in4[k];
          acc = fmaf(w.x, x.x, acc); acc = fmaf(w.y, x.y, acc);
          acc = fmaf(w.z, x.z, acc); acc = fmaf(w.w, x.w, acc);
        }
        float e = __expf(acc);
        ep[o]   = acc > 0.f ? 1.f : e;
        actB[o] = acc > 0.f ? acc : (e - 1.f);
      }
      dense64(S_W1, S_B1, actB, actA, ep + 64);
      dense64(S_W2, S_B2, actA, actB, ep + 128);
      dense64(S_W3, S_B3, actB, actA, ep + 192);
      if (lane < OBS) {
        float acc = sf[S_B4 + lane];
        const float4* w4p = (const float4*)&sf[S_W4 + lane * 68];
        const float4* in4 = (const float4*)actA;
#pragma unroll
        for (int k = 0; k < 16; ++k) {
          float4 w = w4p[k], x = in4[k];
          acc = fmaf(w.x, x.x, acc); acc = fmaf(w.y, x.y, acc);
          acc = fmaf(w.z, x.z, acc); acc = fmaf(w.w, x.w, acc);
        }
        float fy = acc * sSc[lane];
        aug[400 + lane] = yv[lane] - yp[lane] - DTC * fy;  // g
      }

      // ============ Jacobian tangent chain (WMMA f16-in / f32-acc) ============
      v8f D[4][2];
      {  // layer 1 (W1): preload A-frags first (DS in-order: they land while
         // the DS-heavy W0s-based B-build runs), then GEMM with zero waits.
        v16h Af[4][2];
#pragma unroll
        for (int mt = 0; mt < 4; ++mt)
#pragma unroll
          for (int kc = 0; kc < 2; ++kc)
            Af[mt][kc] = *(const v16h*)&fr[FR_W1 + ((mt * 2 + kc) * 32 + lane) * 16];
        v16h Bf[2][2];
#pragma unroll
        for (int kc = 0; kc < 2; ++kc) {
          int kb = 32 * kc + (lo ? 0 : 16);
          float epk[16];
          const float4* e4 = (const float4*)&ep[kb];
#pragma unroll
          for (int q = 0; q < 4; ++q) {
            float4 v = e4[q];
            epk[4 * q] = v.x; epk[4 * q + 1] = v.y;
            epk[4 * q + 2] = v.z; epk[4 * q + 3] = v.w;
          }
#pragma unroll
          for (int nt = 0; nt < 2; ++nt) {
            int n = 16 * nt + nl;
            v16h b;
#pragma unroll
            for (int e = 0; e < 16; ++e)
              b[e] = (_Float16)(epk[e] * sf[S_W0S + (kb + e) * 36 + n]);
            Bf[kc][nt] = b;
          }
        }
#pragma unroll
        for (int mt = 0; mt < 4; ++mt) {
          v8f a0 = zero8(), a1 = zero8();
#pragma unroll
          for (int kc = 0; kc < 2; ++kc) {
            a0 = wmma16(Af[mt][kc], Bf[kc][0], a0);
            a1 = wmma16(Af[mt][kc], Bf[kc][1], a1);
          }
          D[mt][0] = a0; D[mt][1] = a1;
        }
      }
      // layers W2, W3: buildB (kills old D), then batched frag preload, then GEMM
      const int frOff[2] = {FR_W2, FR_W3};
#pragma unroll
      for (int L = 0; L < 2; ++L) {
        v16h Bf[2][2];
        buildB(ep + (L + 1) * WIDTH, D, Bf);
        v16h Af[4][2];
#pragma unroll
        for (int mt = 0; mt < 4; ++mt)
#pragma unroll
          for (int kc = 0; kc < 2; ++kc)
            Af[mt][kc] = *(const v16h*)&fr[frOff[L] + ((mt * 2 + kc) * 32 + lane) * 16];
        v8f nD[4][2];
#pragma unroll
        for (int mt = 0; mt < 4; ++mt) {
          v8f a0 = zero8(), a1 = zero8();
#pragma unroll
          for (int kc = 0; kc < 2; ++kc) {
            a0 = wmma16(Af[mt][kc], Bf[kc][0], a0);
            a1 = wmma16(Af[mt][kc], Bf[kc][1], a1);
          }
          nD[mt][0] = a0; nD[mt][1] = a1;
        }
#pragma unroll
        for (int mt = 0; mt < 4; ++mt) { D[mt][0] = nD[mt][0]; D[mt][1] = nD[mt][1]; }
      }
      {  // output layer W4, then Newton matrix I - dt * diag(scale) * J
        v16h Bf[2][2];
        buildB(ep + 3 * WIDTH, D, Bf);
        v16h Af[2][2];
#pragma unroll
        for (int mt = 0; mt < 2; ++mt)
#pragma unroll
          for (int kc = 0; kc < 2; ++kc)
            Af[mt][kc] = *(const v16h*)&fr[FR_W4 + ((mt * 2 + kc) * 32 + lane) * 16];
#pragma unroll
        for (int mt = 0; mt < 2; ++mt) {
          v8f a0 = zero8(), a1 = zero8();
#pragma unroll
          for (int kc = 0; kc < 2; ++kc) {
            a0 = wmma16(Af[mt][kc], Bf[kc][0], a0);
            a1 = wmma16(Af[mt][kc], Bf[kc][1], a1);
          }
#pragma unroll
          for (int nt = 0; nt < 2; ++nt) {
            v8f F = nt ? a1 : a0;
#pragma unroll
            for (int r = 0; r < 8; ++r) {
              int m = 16 * mt + roff + r;
              int n = 16 * nt + nl;
              if (m < OBS && n < OBS)
                aug[n * OBS + m] = ((m == n) ? 1.f : 0.f) - DTC * (sSc[m] * F[r]);
            }
          }
        }
      }

      // ==== 20x21 Gauss-Jordan, lanes = columns, v_readlane broadcasts ====
      float c[OBS];
      {
        int col = (lane < 21) ? lane : 0;
#pragma unroll
        for (int r = 0; r < OBS; ++r) c[r] = aug[col * OBS + r];
      }
#pragma unroll
      for (int k = 0; k < OBS; ++k) {
        float piv  = rdlane(c[k], k);
        float pinv = 1.0f / piv;
        float oldk = c[k];
#pragma unroll
        for (int r = 0; r < OBS; ++r) {
          if (r == k) continue;
          float fct = rdlane(c[r], k) * pinv;
          c[r] = fmaf(-fct, oldk, c[r]);
        }
        c[k] = oldk * pinv;
      }
      if (lane == 20) {
#pragma unroll
        for (int r = 0; r < OBS; ++r) aug[400 + r] = c[r];
      }
      if (lane < OBS) yv[lane] -= aug[400 + lane];  // Newton update
    }  // newton

    if (lane < OBS) {
      float v = yv[lane];
      yp[lane] = v;
      out[(size_t)(t + 1) * BATCHN * OBS + (size_t)traj * OBS + lane] = v;
    }
  }  // steps
}

extern "C" void kernel_launch(void* const* d_in, const int* in_sizes, int n_in,
                              void* d_out, int out_size, void* d_ws, size_t ws_size,
                              hipStream_t stream) {
  (void)in_sizes; (void)n_in; (void)out_size; (void)d_ws; (void)ws_size;
  const float* y0 = (const float*)d_in[0];
  const float* W0 = (const float*)d_in[1];  const float* b0 = (const float*)d_in[2];
  const float* W1 = (const float*)d_in[3];  const float* b1 = (const float*)d_in[4];
  const float* W2 = (const float*)d_in[5];  const float* b2 = (const float*)d_in[6];
  const float* W3 = (const float*)d_in[7];  const float* b3 = (const float*)d_in[8];
  const float* W4 = (const float*)d_in[9];  const float* b4 = (const float*)d_in[10];
  const float* scale = (const float*)d_in[11];
  const int*   nsp   = (const int*)d_in[12];
  float* out = (float*)d_out;

  (void)hipFuncSetAttribute((const void*)stiff_node_kernel,
                            hipFuncAttributeMaxDynamicSharedMemorySize, LDS_BYTES);
  dim3 grid(BATCHN / WAVES_PER_WG);  // 512 WGs x 8 waves = 4096 trajectories
  hipLaunchKernelGGL(stiff_node_kernel, grid, dim3(BLOCKSZ), LDS_BYTES, stream,
                     y0, W0, b0, W1, b1, W2, b2, W3, b3, W4, b4, scale, nsp, out);
}
